// EmformerEncoder_72834055405980
// MI455X (gfx1250) — matrix operand order repaired and checked by reading, not compiled, not run on hardware
//
#include <hip/hip_runtime.h>
#include <hip/hip_bf16.h>

// ---------------------------------------------------------------------------
// Emformer encoder forward for MI455X (gfx1250, wave32, WMMA f32_16x16x32_f16)
// ---------------------------------------------------------------------------

#define B_   4
#define T_   4096
#define TN   1024          // T_/STACK
#define R_   248           // (num_chunks-1)*RIGHT = 31*8
#define S_   1272          // R_ + TN
#define D_   512
#define H_   8
#define DK   64
#define F_   2048
#define M_   (B_*S_)       // 5088 rows, divisible by 16 (318 tiles)

typedef __attribute__((ext_vector_type(16))) _Float16     v16h;
typedef __attribute__((ext_vector_type(8)))  float        v8f;
typedef __attribute__((ext_vector_type(4)))  unsigned int v4u;

union AFrag { v16h v; _Float16 e[16]; v4u q[2]; };
union CFrag { v8f  v; float    e[8];  };

static __device__ __forceinline__ v8f wmma_f16(v16h a, v16h b, v8f c) {
  return __builtin_amdgcn_wmma_f32_16x16x32_f16(false, a, false, b, (short)0, c, false, false);
}

// A-matrix 16x32 f16 fragment (ISA 7.12.2): lane m = lane&15, h = lane>>4.
// Lane data = halves [k0+8h .. k0+8h+7] and [k0+16+8h .. k0+16+8h+7]:
// two contiguous 16-byte chunks -> two b128 loads, no repacking.
static __device__ __forceinline__ v16h load_a_row(const _Float16* __restrict__ rowbase,
                                                  int k0, int h) {
  AFrag a;
  a.q[0] = *(const v4u*)(rowbase + k0 + 8*h);
  a.q[1] = *(const v4u*)(rowbase + k0 + 16 + 8*h);
  return a.v;
}

// B-matrix 32x16 f16 fragment: lane n fixed (rowbase = &B[n][.] of an [N,K]
// layout). Lane data = halves [k0+8h .. k0+8h+7] and [k0+16+8h .. k0+16+8h+7]:
// two contiguous 16-byte chunks -> two b128 loads, no repacking.
static __device__ __forceinline__ v16h load_b_row(const _Float16* __restrict__ rowbase,
                                                  int k0, int h) {
  AFrag b;
  const _Float16* p = rowbase + k0 + 8*h;
  b.q[0] = *(const v4u*)(p);
  b.q[1] = *(const v4u*)(p + 16);
  return b.v;
}

// Element-wise clamped variant (only used on the final key tile).
static __device__ __forceinline__ v16h load_b_row_clamped(const _Float16* __restrict__ rowbase,
                                                          int k0, int h, int kmax) {
  AFrag b;
#pragma unroll
  for (int i = 0; i < 16; ++i) {
    int k = k0 + 8*h + (i < 8 ? i : 8 + i);
    b.e[i] = rowbase[k > kmax ? kmax : k];
  }
  return b.v;
}

// ---------------------------------------------------------------------------
// Frontend: x[b,s,:] = stack4( audio[b,:,:]^T @ W_lin + b_lin ) with right-
// context gather folded in (row s<R reads body index ri[s]).
// ---------------------------------------------------------------------------
__global__ __launch_bounds__(256) void frontend_kernel(const float* __restrict__ audio,
                                                       const float* __restrict__ Wl,
                                                       const float* __restrict__ bl,
                                                       float* __restrict__ xout) {
  __shared__ float Ws[80*128];
  __shared__ float As[4*80];
  int row = blockIdx.x;                 // b*S_ + s
  int b = row / S_, s = row % S_;
  int tt = (s < R_) ? (32*((s>>3)+1) + (s&7)) : (s - R_);
  int tid = threadIdx.x;
  for (int i = tid; i < 80*128; i += 256) Ws[i] = Wl[i];
  for (int i = tid; i < 320; i += 256) {
    int k = i / 80, d = i % 80;
    As[i] = audio[((size_t)(b*80 + d))*T_ + 4*tt + k];
  }
  __syncthreads();
#pragma unroll
  for (int rep = 0; rep < 2; ++rep) {
    int fo = tid + rep*256;             // 0..511
    int kk = fo >> 7, f = fo & 127;
    float sum = bl[f];
#pragma unroll 4
    for (int d = 0; d < 80; ++d) sum += As[kk*80 + d] * Ws[d*128 + f];
    xout[(size_t)row * D_ + fo] = sum;
  }
}

// ---------------------------------------------------------------------------
// Weight convert + transpose: Wt[n*K+k] = (f16) W[k*N+n]
// ---------------------------------------------------------------------------
__global__ __launch_bounds__(256) void tconv_kernel(const float* __restrict__ W,
                                                    _Float16* __restrict__ Wt,
                                                    int K, int N) {
  int idx = blockIdx.x * 256 + threadIdx.x;
  if (idx < K * N) {
    int n = idx % N, k = idx / N;
    Wt[(size_t)n * K + k] = (_Float16)W[idx];
  }
}

// ---------------------------------------------------------------------------
// LayerNorm (one block per 512-wide row); optional f32 + f16 outputs.
// ---------------------------------------------------------------------------
__global__ __launch_bounds__(256) void ln_kernel(const float* __restrict__ in,
                                                 const float* __restrict__ g,
                                                 const float* __restrict__ bta,
                                                 float* __restrict__ out32,
                                                 _Float16* __restrict__ out16) {
  __shared__ float red[256];
  int row = blockIdx.x, tid = threadIdx.x;
  const float* x = in + (size_t)row * D_;
  float v0 = x[tid], v1 = x[tid + 256];
  red[tid] = v0 + v1; __syncthreads();
  for (int st = 128; st > 0; st >>= 1) { if (tid < st) red[tid] += red[tid+st]; __syncthreads(); }
  float mean = red[0] * (1.0f/512.0f); __syncthreads();
  float d0 = v0 - mean, d1 = v1 - mean;
  red[tid] = d0*d0 + d1*d1; __syncthreads();
  for (int st = 128; st > 0; st >>= 1) { if (tid < st) red[tid] += red[tid+st]; __syncthreads(); }
  float r = rsqrtf(red[0] * (1.0f/512.0f) + 1e-5f);
  float o0 = d0 * r * g[tid]       + bta[tid];
  float o1 = d1 * r * g[tid + 256] + bta[tid + 256];
  size_t base = (size_t)row * D_;
  if (out32) { out32[base + tid] = o0; out32[base + tid + 256] = o1; }
  if (out16) { out16[base + tid] = (_Float16)o0; out16[base + tid + 256] = (_Float16)o1; }
}

// y = a + b (stored to ybuf, may alias b), then LayerNorm(y) -> f16
__global__ __launch_bounds__(256) void ln_add_kernel(const float* __restrict__ a,
                                                     const float* __restrict__ bb,
                                                     const float* __restrict__ g,
                                                     const float* __restrict__ bta,
                                                     float* __restrict__ ybuf,
                                                     _Float16* __restrict__ out16) {
  __shared__ float red[256];
  int row = blockIdx.x, tid = threadIdx.x;
  size_t base = (size_t)row * D_;
  float v0 = a[base + tid]       + bb[base + tid];
  float v1 = a[base + tid + 256] + bb[base + tid + 256];
  ybuf[base + tid] = v0; ybuf[base + tid + 256] = v1;
  red[tid] = v0 + v1; __syncthreads();
  for (int st = 128; st > 0; st >>= 1) { if (tid < st) red[tid] += red[tid+st]; __syncthreads(); }
  float mean = red[0] * (1.0f/512.0f); __syncthreads();
  float d0 = v0 - mean, d1 = v1 - mean;
  red[tid] = d0*d0 + d1*d1; __syncthreads();
  for (int st = 128; st > 0; st >>= 1) { if (tid < st) red[tid] += red[tid+st]; __syncthreads(); }
  float r = rsqrtf(red[0] * (1.0f/512.0f) + 1e-5f);
  out16[base + tid]       = (_Float16)(d0 * r * g[tid]       + bta[tid]);
  out16[base + tid + 256] = (_Float16)(d1 * r * g[tid + 256] + bta[tid + 256]);
}

// ---------------------------------------------------------------------------
// Generic WMMA GEMM: C[M,N] = A16[M,K] @ Wt16[N,K]^T + bias[N], epilogues:
//  EPI 0: f16 out [M,N]              (Q,K projections)
//  EPI 1: f16 out transposed to [B,H,DK,S]   (V projection)
//  EPI 2: f16 out [M,N]              (FFN mid)
//  EPI 3: f32 out [M,N] + residual   (FFN out + attn_out residual)
// Each wave computes a 16x64 tile (A fragment reused across 4 N-tiles).
// ---------------------------------------------------------------------------
template <int EPI>
__global__ __launch_bounds__(256) void gemm_ws(const _Float16* __restrict__ A,
                                               const _Float16* __restrict__ Bt,
                                               const float* __restrict__ bias,
                                               const float* __restrict__ res,
                                               _Float16* __restrict__ out16,
                                               float* __restrict__ out32,
                                               int M, int N, int K) {
  int lane = threadIdx.x & 31;
  int wave = threadIdx.x >> 5;
  int tile = blockIdx.x * 8 + wave;
  int nt = N >> 6;
  int m0 = (tile / nt) << 4;
  int n0 = (tile % nt) << 6;
  int h = lane >> 4, ln = lane & 15;
  const _Float16* arow = A + (size_t)(m0 + ln) * K;
  CFrag acc[4];
#pragma unroll
  for (int t = 0; t < 4; ++t)
#pragma unroll
    for (int e = 0; e < 8; ++e) acc[t].e[e] = 0.0f;

  for (int k0 = 0; k0 < K; k0 += 32) {
    v16h a = load_a_row(arow, k0, h);
#pragma unroll
    for (int t = 0; t < 4; ++t) {
      const _Float16* brow = Bt + (size_t)(n0 + t*16 + ln) * K;
      v16h b = load_b_row(brow, k0, h);
      acc[t].v = wmma_f16(a, b, acc[t].v);
    }
  }
#pragma unroll
  for (int t = 0; t < 4; ++t) {
    int n = n0 + t*16 + ln;
    float bi = bias[n];
#pragma unroll
    for (int i = 0; i < 8; ++i) {
      int m = m0 + i + 8*h;
      float val = acc[t].e[i] + bi;
      if (EPI == 0 || EPI == 2) {
        out16[(size_t)m * N + n] = (_Float16)val;
      } else if (EPI == 1) {
        int b_ = m / S_, s_ = m % S_;
        int hd = n >> 6, dk = n & 63;
        out16[((size_t)((b_*H_ + hd)*DK + dk))*S_ + s_] = (_Float16)val;
      } else {
        out32[(size_t)m * N + n] = val + res[(size_t)m * N + n];
      }
    }
  }
}

// ---------------------------------------------------------------------------
// Emformer mask predicate, derived analytically from create_mask():
//   L = ceil(len/4), pb = #right-indexes < L, ri[j] = 32*(j/8+1)+j%8
// ---------------------------------------------------------------------------
static __device__ __forceinline__ bool allowed(int row, int col, int L, int pb) {
  if (row >= S_ || col >= S_) return false;
  if (row < R_) {                       // right-context summary row
    if (row >= pb) return false;
    if (col < R_) return col == row;    // diag
    return ((col - R_) >> 5) == (row >> 3);
  }
  int t = row - R_;
  if (t >= L) return false;
  int i = t >> 5;
  if (col < R_) return ((col >> 3) == i) && (col < pb);
  int u = col - R_;
  if (u >= L) return false;
  int lo = (i > 0) ? ((i - 1) << 5) : 0;
  return (u >= lo) && (u < ((i + 1) << 5));
}

// ---------------------------------------------------------------------------
// Flash attention: one wave per (b, head, 16-query tile); 32-key tiles.
// Scores via 4 WMMAs, online softmax, P re-layout through LDS, P@V via 4 WMMAs.
// ---------------------------------------------------------------------------
__global__ __launch_bounds__(32) void attn_kernel(const _Float16* __restrict__ q16,
                                                  const _Float16* __restrict__ k16,
                                                  const _Float16* __restrict__ vt16,
                                                  const int* __restrict__ length,
                                                  float* __restrict__ attn32) {
  __shared__ _Float16 pl[16 * 32];
  const int QT = 80;                    // ceil(S_/16)
  int blk = blockIdx.x;
  int qt = blk % QT;
  int rem = blk / QT;
  int head = rem % H_;
  int b = rem / H_;
  int lane = threadIdx.x;
  int h = lane >> 4, ln = lane & 15;
  const float NEGINF = -__builtin_inff();

  int L = (length[b] + 3) >> 2;
  int pb = 0;
#pragma unroll
  for (int i = 1; i <= 31; ++i) {
    int c = L - 32*i; c = c < 0 ? 0 : (c > 8 ? 8 : c); pb += c;
  }

  int q0 = qt * 16;
  int qrow = q0 + ln; if (qrow >= S_) qrow = S_ - 1;
  const _Float16* qbase = q16 + ((size_t)(b * S_ + qrow)) * D_ + head * DK;
  v16h qa0 = load_a_row(qbase, 0, h);
  v16h qa1 = load_a_row(qbase, 32, h);

  CFrag acc[4];
#pragma unroll
  for (int t = 0; t < 4; ++t)
#pragma unroll
    for (int e = 0; e < 8; ++e) acc[t].e[e] = 0.0f;
  float rowmax[8], rowsum[8];
#pragma unroll
  for (int i = 0; i < 8; ++i) { rowmax[i] = NEGINF; rowsum[i] = 0.0f; }

  CFrag zf;
#pragma unroll
  for (int e = 0; e < 8; ++e) zf.e[e] = 0.0f;

  for (int kt = 0; kt < 40; ++kt) {
    int key0 = kt * 32;
    CFrag s[2];
#pragma unroll
    for (int sub = 0; sub < 2; ++sub) {
      int key = key0 + sub*16 + ln; if (key >= S_) key = S_ - 1;
      const _Float16* kbase = k16 + ((size_t)(b * S_ + key)) * D_ + head * DK;
      v16h kb0 = load_b_row(kbase, 0, h);
      v16h kb1 = load_b_row(kbase, 32, h);
      v8f c = wmma_f16(qa0, kb0, zf.v);
      s[sub].v = wmma_f16(qa1, kb1, c);
    }
    float p0[8], p1[8];
#pragma unroll
    for (int i = 0; i < 8; ++i) {
      int row = q0 + i + 8*h;
      float v0 = s[0].e[i] * 0.125f;     // 1/sqrt(64)
      float v1 = s[1].e[i] * 0.125f;
      if (!allowed(row, key0 + ln,      L, pb)) v0 = NEGINF;
      if (!allowed(row, key0 + 16 + ln, L, pb)) v1 = NEGINF;
      float tm = fmaxf(v0, v1);
#pragma unroll
      for (int mk = 1; mk < 16; mk <<= 1) tm = fmaxf(tm, __shfl_xor(tm, mk, 32));
      float nm = fmaxf(rowmax[i], tm);
      float sc = (rowmax[i] == NEGINF) ? 0.0f : __expf(rowmax[i] - nm);
      float e0 = (v0 == NEGINF) ? 0.0f : __expf(v0 - nm);
      float e1 = (v1 == NEGINF) ? 0.0f : __expf(v1 - nm);
      float ts = e0 + e1;
#pragma unroll
      for (int mk = 1; mk < 16; mk <<= 1) ts += __shfl_xor(ts, mk, 32);
      rowsum[i] = rowsum[i] * sc + ts;
      rowmax[i] = nm;
      p0[i] = e0; p1[i] = e1;
#pragma unroll
      for (int t = 0; t < 4; ++t) acc[t].e[i] *= sc;
    }
    __syncthreads();                    // protect previous pa loads
#pragma unroll
    for (int i = 0; i < 8; ++i) {
      pl[(i + 8*h) * 32 + ln]      = (_Float16)p0[i];
      pl[(i + 8*h) * 32 + 16 + ln] = (_Float16)p1[i];
    }
    __syncthreads();
    v16h pa = load_a_row(pl + ln * 32, 0, h);   // P as A-fragment (16x32)
    bool full = (key0 + 32 <= S_);
#pragma unroll
    for (int t = 0; t < 4; ++t) {
      const _Float16* vbase = vt16 + ((size_t)((b*H_ + head)*DK + (t*16 + ln))) * S_;
      v16h vb = full ? load_b_row(vbase, key0, h)
                     : load_b_row_clamped(vbase, key0, h, S_ - 1);
      acc[t].v = wmma_f16(pa, vb, acc[t].v);
    }
  }
#pragma unroll
  for (int i = 0; i < 8; ++i) {
    int row = q0 + i + 8*h;
    if (row < S_) {
      float inv = (rowsum[i] > 0.0f) ? (1.0f / rowsum[i]) : 0.0f;
#pragma unroll
      for (int t = 0; t < 4; ++t)
        attn32[((size_t)(b * S_ + row)) * D_ + head * DK + t*16 + ln] = acc[t].e[i] * inv;
    }
  }
}

// ---------------------------------------------------------------------------
// Output slice + transpose: out[b,d,t] = x[b, R_+t, d]
// ---------------------------------------------------------------------------
__global__ __launch_bounds__(256) void outtr_kernel(const float* __restrict__ x,
                                                    float* __restrict__ out) {
  int idx = blockIdx.x * 256 + threadIdx.x;
  if (idx < B_ * D_ * TN) {
    int t = idx % TN;
    int rest = idx / TN;
    int d = rest % D_;
    int b = rest / D_;
    out[idx] = x[((size_t)(b * S_ + R_ + t)) * D_ + d];
  }
}

// ---------------------------------------------------------------------------
extern "C" void kernel_launch(void* const* d_in, const int* in_sizes, int n_in,
                              void* d_out, int out_size, void* d_ws, size_t ws_size,
                              hipStream_t stream) {
  (void)in_sizes; (void)n_in; (void)out_size; (void)ws_size;
  const float* audio   = (const float*)d_in[0];
  const int*   length  = (const int*)  d_in[1];
  const float* W_lin   = (const float*)d_in[2];
  const float* b_lin   = (const float*)d_in[3];
  const float* ln_in_g = (const float*)d_in[4];
  const float* ln_in_b = (const float*)d_in[5];
  const float* Wq      = (const float*)d_in[6];
  const float* bq      = (const float*)d_in[7];
  const float* Wk      = (const float*)d_in[8];
  const float* bk      = (const float*)d_in[9];
  const float* Wv      = (const float*)d_in[10];
  const float* bv      = (const float*)d_in[11];
  const float* ln1_g   = (const float*)d_in[12];
  const float* ln1_b   = (const float*)d_in[13];
  const float* W1      = (const float*)d_in[14];
  const float* bf1     = (const float*)d_in[15];
  const float* W2      = (const float*)d_in[16];
  const float* bf2     = (const float*)d_in[17];
  const float* ln2_g   = (const float*)d_in[18];
  const float* ln2_b   = (const float*)d_in[19];

  // workspace carve-up
  size_t off = 0;
  auto alloc = [&](size_t bytes) -> void* {
    off = (off + 255) & ~(size_t)255;
    void* p = (char*)d_ws + off;
    off += bytes;
    return p;
  };
  float*     xbuf   = (float*)    alloc((size_t)M_ * D_ * 4);
  float*     h32    = (float*)    alloc((size_t)M_ * D_ * 4);   // LN out, then y
  _Float16*  h16    = (_Float16*) alloc((size_t)M_ * D_ * 2);   // LN out f16 / z16
  _Float16*  q16    = (_Float16*) alloc((size_t)M_ * D_ * 2);
  _Float16*  k16    = (_Float16*) alloc((size_t)M_ * D_ * 2);
  _Float16*  vt16   = (_Float16*) alloc((size_t)M_ * D_ * 2);   // [B,H,DK,S]
  float*     attn32 = (float*)    alloc((size_t)M_ * D_ * 4);   // attn, then y2
  _Float16*  mid16  = (_Float16*) alloc((size_t)M_ * F_ * 2);
  _Float16*  wqt[4], *wkt[4], *wvt[4], *w1t[4], *w2t[4];
  for (int i = 0; i < 4; ++i) {
    wqt[i] = (_Float16*)alloc((size_t)D_ * D_ * 2);
    wkt[i] = (_Float16*)alloc((size_t)D_ * D_ * 2);
    wvt[i] = (_Float16*)alloc((size_t)D_ * D_ * 2);
    w1t[i] = (_Float16*)alloc((size_t)D_ * F_ * 2);
    w2t[i] = (_Float16*)alloc((size_t)F_ * D_ * 2);
  }

  // weight convert/transpose (once per launch)
  for (int i = 0; i < 4; ++i) {
    tconv_kernel<<<(D_*D_+255)/256, 256, 0, stream>>>(Wq + (size_t)i*D_*D_, wqt[i], D_, D_);
    tconv_kernel<<<(D_*D_+255)/256, 256, 0, stream>>>(Wk + (size_t)i*D_*D_, wkt[i], D_, D_);
    tconv_kernel<<<(D_*D_+255)/256, 256, 0, stream>>>(Wv + (size_t)i*D_*D_, wvt[i], D_, D_);
    tconv_kernel<<<(D_*F_+255)/256, 256, 0, stream>>>(W1 + (size_t)i*D_*F_, w1t[i], D_, F_);
    tconv_kernel<<<(F_*D_+255)/256, 256, 0, stream>>>(W2 + (size_t)i*F_*D_, w2t[i], F_, D_);
  }

  // frontend + right-context gather -> xbuf
  frontend_kernel<<<M_, 256, 0, stream>>>(audio, W_lin, b_lin, xbuf);

  const int gQKV  = (M_/16) * (D_/64) / 8;   // 318
  const int gFFN1 = (M_/16) * (F_/64) / 8;   // 1272
  const int gATT  = B_ * H_ * 80;            // 2560

  for (int i = 0; i < 4; ++i) {
    // LN_in -> h32 (f32) + h16 (f16)
    ln_kernel<<<M_, 256, 0, stream>>>(xbuf, ln_in_g + i*D_, ln_in_b + i*D_, h32, h16);
    // Q,K,V projections
    gemm_ws<0><<<gQKV, 256, 0, stream>>>(h16, wqt[i], bq + i*D_, nullptr, q16, nullptr, M_, D_, D_);
    gemm_ws<0><<<gQKV, 256, 0, stream>>>(h16, wkt[i], bk + i*D_, nullptr, k16, nullptr, M_, D_, D_);
    gemm_ws<1><<<gQKV, 256, 0, stream>>>(h16, wvt[i], bv + i*D_, nullptr, vt16, nullptr, M_, D_, D_);
    // masked flash attention
    attn_kernel<<<gATT, 32, 0, stream>>>(q16, k16, vt16, length, attn32);
    // y = attn + h ; LN1(y) -> h16 (z16); y kept in h32
    ln_add_kernel<<<M_, 256, 0, stream>>>(attn32, h32, ln1_g + i*D_, ln1_b + i*D_, h32, h16);
    // FFN: mid = z @ W1 + bf1 ; y2 = mid @ W2 + bf2 + y
    gemm_ws<2><<<gFFN1, 256, 0, stream>>>(h16, w1t[i], bf1 + i*F_, nullptr, mid16, nullptr, M_, F_, D_);
    gemm_ws<3><<<gQKV, 256, 0, stream>>>(mid16, w2t[i], bf2 + i*D_, h32, nullptr, attn32, M_, D_, F_);
    // LN2 -> next layer input
    ln_kernel<<<M_, 256, 0, stream>>>(attn32, ln2_g + i*D_, ln2_b + i*D_, xbuf, nullptr);
  }

  outtr_kernel<<<(B_*D_*TN + 255)/256, 256, 0, stream>>>(xbuf, (float*)d_out);
}
